// LeWinTransformerBlock_21053929685593
// MI455X (gfx1250) — compile-verified, matrix operand order checked by reading, not run on hardware
//
#include <hip/hip_runtime.h>

typedef __bf16 bf16;
typedef __attribute__((ext_vector_type(16))) __bf16 bf16x16;
typedef __attribute__((ext_vector_type(8)))  __bf16 bf16x8;
typedef __attribute__((ext_vector_type(8)))  float  floatx8;

// ---------- helpers ----------
__device__ __forceinline__ bf16 f2bf(float f) {
    unsigned u = __float_as_uint(f);
    u += 0x7FFFu + ((u >> 16) & 1u);            // round-to-nearest-even
    unsigned short s = (unsigned short)(u >> 16);
    bf16 r; __builtin_memcpy(&r, &s, 2);
    return r;
}
__device__ __forceinline__ float bf2f(bf16 b) {
    unsigned short s; __builtin_memcpy(&s, &b, 2);
    unsigned u = ((unsigned)s) << 16;
    return __uint_as_float(u);
}
__device__ __forceinline__ float geluf(float v) {
    return 0.5f * v * (1.0f + erff(v * 0.70710678118654752f));
}

// CDNA5 async global->LDS copy (16B per enabled lane), tracked by ASYNCcnt.
// vdst VGPR carries the LDS byte offset; vaddr is the 64-bit global address.
__device__ __forceinline__ void async_load_b128(unsigned lds_off, const void* gaddr) {
    asm volatile("global_load_async_to_lds_b128 %0, %1, off"
                 :: "v"(lds_off), "v"(gaddr) : "memory");
}
__device__ __forceinline__ void wait_asynccnt0() {
    asm volatile("s_wait_asynccnt 0x0" ::: "memory");
}

// bf16 16x16x32 WMMA fragment load from row-major storage.
// A-matrix 16x32 layout (ISA 7.12.2): lane = {r = lane&15, lh = lane>>4};
// elements j<8  -> K = kbase + 8*lh + j        (16 contiguous bytes)
// elements j>=8 -> K = kbase + 16 + 8*lh + j-8 (16 contiguous bytes)
// B operand uses the mirrored layout, so loading from the transposed (NxK)
// matrix with the same code is correct (lane&15 = N column).
__device__ __forceinline__ bf16x16 load_frag(const bf16* base, int row, int pitch,
                                             int kbase, int lane) {
    int r = lane & 15, lh = lane >> 4;
    const bf16* p = base + (size_t)(row + r) * pitch + kbase + lh * 8;
    union { bf16x16 f; bf16x8 h[2]; } u;
    u.h[0] = *reinterpret_cast<const bf16x8*>(p);
    u.h[1] = *reinterpret_cast<const bf16x8*>(p + 16);
    return u.f;
}

__device__ __forceinline__ floatx8 wmma_bf16(bf16x16 a, bf16x16 b, floatx8 c) {
    return __builtin_amdgcn_wmma_f32_16x16x32_bf16(false, a, false, b, (short)0, c,
                                                   false, false);
}

// ---------- K0: weight transpose + cast: src[K][N] f32 -> dst[N][K] bf16 ----------
__global__ void k_wt(const float* __restrict__ src, bf16* __restrict__ dst, int K, int N) {
    int i = blockIdx.x * 256 + threadIdx.x;
    if (i >= K * N) return;
    int n = i / K, k = i - n * K;
    dst[(size_t)n * K + k] = f2bf(src[(size_t)k * N + n]);
}

// ---------- K1: LN1 + cyclic shift (-4,-4) + window partition -> xw bf16 ----------
__global__ void k_ln_shift_part(const float* __restrict__ x, const float* __restrict__ g,
                                const float* __restrict__ be, bf16* __restrict__ xw) {
    int wave = threadIdx.x >> 5, lane = threadIdx.x & 31;
    int t = blockIdx.x * 8 + wave;                 // token 0..65535
    int b = t >> 14, l = t & 16383;
    const float* row = x + (size_t)t * 256 + lane * 8;
    float v[8];
    float4 p0 = *reinterpret_cast<const float4*>(row);
    float4 p1 = *reinterpret_cast<const float4*>(row + 4);
    v[0]=p0.x; v[1]=p0.y; v[2]=p0.z; v[3]=p0.w;
    v[4]=p1.x; v[5]=p1.y; v[6]=p1.z; v[7]=p1.w;
    float s = 0.f, s2 = 0.f;
    #pragma unroll
    for (int j = 0; j < 8; ++j) { s += v[j]; s2 += v[j]*v[j]; }
    #pragma unroll
    for (int m = 16; m >= 1; m >>= 1) {
        s  += __shfl_xor(s,  m, 32);
        s2 += __shfl_xor(s2, m, 32);
    }
    float mu = s * (1.f/256.f);
    float var = s2 * (1.f/256.f) - mu * mu;
    float rstd = rsqrtf(var + 1e-5f);
    int h = l >> 7, w = l & 127;
    int hs = (h + 124) & 127, wsx = (w + 124) & 127;   // shifted position
    int wy = hs >> 3, wx = wsx >> 3;
    int win = (b << 8) + wy * 16 + wx;
    int n = (hs & 7) * 8 + (wsx & 7);
    union { bf16x8 vec; bf16 e[8]; } o;
    #pragma unroll
    for (int j = 0; j < 8; ++j)
        o.e[j] = f2bf((v[j] - mu) * rstd * g[lane*8+j] + be[lane*8+j]);
    *reinterpret_cast<bf16x8*>(xw + ((size_t)win * 64 + n) * 256 + lane * 8) = o.vec;
}

// ---------- K2: QKV projection per window (WMMA) ----------
__global__ void k_qkv(const bf16* __restrict__ xw, const bf16* __restrict__ wqT,
                      const bf16* __restrict__ wkvT, const float* __restrict__ bq,
                      const float* __restrict__ bkv, bf16* __restrict__ q,
                      bf16* __restrict__ kk, bf16* __restrict__ vT) {
    int wave = threadIdx.x >> 5, lane = threadIdx.x & 31;
    int win = blockIdx.x / 6, sb = blockIdx.x - win * 6;
    int nbase = (sb * 8 + wave) * 16;              // 0..767
    const bf16* A  = xw + (size_t)win * 64 * 256;
    const bf16* BT = (nbase < 256) ? (wqT + (size_t)nbase * 256)
                                   : (wkvT + (size_t)(nbase - 256) * 256);
    floatx8 acc[4] = {};
    for (int kb = 0; kb < 256; kb += 32) {
        bf16x16 bfr = load_frag(BT, 0, 256, kb, lane);
        #pragma unroll
        for (int mt = 0; mt < 4; ++mt)
            acc[mt] = wmma_bf16(load_frag(A, mt*16, 256, kb, lane), bfr, acc[mt]);
    }
    int cn = lane & 15, lh = lane >> 4;
    int c = nbase + cn;
    if (c < 256) {                                  // Q, fold in 1/sqrt(hd)
        float bias = bq[c];
        #pragma unroll
        for (int mt = 0; mt < 4; ++mt)
            #pragma unroll
            for (int e = 0; e < 8; ++e) {
                int n = mt*16 + e + 8*lh;
                q[(size_t)win*16384 + n*256 + c] =
                    f2bf((acc[mt][e] + bias) * 0.17677669529663687f);
            }
    } else if (c < 512) {                           // K row-major
        int ck = c - 256; float bias = bkv[ck];
        #pragma unroll
        for (int mt = 0; mt < 4; ++mt)
            #pragma unroll
            for (int e = 0; e < 8; ++e) {
                int n = mt*16 + e + 8*lh;
                kk[(size_t)win*16384 + n*256 + ck] = f2bf(acc[mt][e] + bias);
            }
    } else {                                        // V transposed [head][d][m]
        int cv = c - 512; float bias = bkv[c - 256];
        int head = cv >> 5, dl = cv & 31;
        #pragma unroll
        for (int mt = 0; mt < 4; ++mt) {
            union { bf16x8 vec; bf16 e[8]; } o;
            #pragma unroll
            for (int e = 0; e < 8; ++e) o.e[e] = f2bf(acc[mt][e] + bias);
            int m0 = mt*16 + 8*lh;
            *reinterpret_cast<bf16x8*>(
                vT + ((size_t)(win*8 + head)*32 + dl)*64 + m0) = o.vec;
        }
    }
}

// ---------- K3: windowed attention, one wave per (window, head) ----------
__global__ void k_attn(const bf16* __restrict__ q, const bf16* __restrict__ kk,
                       const bf16* __restrict__ vT, const float* __restrict__ rpb,
                       bf16* __restrict__ o) {
    __shared__ bf16  att[64 * 64];
    __shared__ float rpb_s[225 * 8];
    for (int i = threadIdx.x; i < 1800; i += 32) rpb_s[i] = rpb[i];
    __syncthreads();
    int lane = threadIdx.x & 31;
    int head = blockIdx.x & 7, win = blockIdx.x >> 3;
    int cn = lane & 15, lh = lane >> 4;

    const bf16* qb = q  + (size_t)win*16384 + head*32;   // [64][32], pitch 256
    const bf16* kb = kk + (size_t)win*16384 + head*32;
    bf16x16 af[4], bfr[4];
    #pragma unroll
    for (int mt = 0; mt < 4; ++mt) af[mt]  = load_frag(qb, mt*16, 256, 0, lane);
    #pragma unroll
    for (int nt = 0; nt < 4; ++nt) bfr[nt] = load_frag(kb, nt*16, 256, 0, lane);
    floatx8 acc[4][4];
    #pragma unroll
    for (int mt = 0; mt < 4; ++mt)
        #pragma unroll
        for (int nt = 0; nt < 4; ++nt) {
            floatx8 z = {};
            acc[mt][nt] = wmma_bf16(af[mt], bfr[nt], z);
        }

    int wy = (win & 255) >> 4, wx = win & 15;
    bool edge = (wy == 15) || (wx == 15);
    #pragma unroll
    for (int mt = 0; mt < 4; ++mt)
        #pragma unroll
        for (int nt = 0; nt < 4; ++nt)
            #pragma unroll
            for (int e = 0; e < 8; ++e) {
                int n = mt*16 + e + 8*lh, m = nt*16 + cn;
                int ny = n >> 3, nx = n & 7, my = m >> 3, mx = m & 7;
                float bias = rpb_s[((ny - my + 7)*15 + (nx - mx + 7))*8 + head];
                if (edge) {
                    int hn = wy*8+ny, wn = wx*8+nx, hm = wy*8+my, wm = wx*8+mx;
                    int idn = (hn<120?0:(hn<124?1:2))*3 + (wn<120?0:(wn<124?1:2));
                    int idm = (hm<120?0:(hm<124?1:2))*3 + (wm<120?0:(wm<124?1:2));
                    if (idn != idm) bias -= 100.f;
                }
                acc[mt][nt][e] += bias;
            }
    // softmax over m (16 lanes in a half-wave x 4 N-tiles)
    #pragma unroll
    for (int mt = 0; mt < 4; ++mt)
        #pragma unroll
        for (int e = 0; e < 8; ++e) {
            float mx = acc[mt][0][e];
            #pragma unroll
            for (int nt = 1; nt < 4; ++nt) mx = fmaxf(mx, acc[mt][nt][e]);
            #pragma unroll
            for (int msk = 8; msk >= 1; msk >>= 1)
                mx = fmaxf(mx, __shfl_xor(mx, msk, 32));
            float sum = 0.f;
            #pragma unroll
            for (int nt = 0; nt < 4; ++nt) {
                float ev = __expf(acc[mt][nt][e] - mx);
                acc[mt][nt][e] = ev; sum += ev;
            }
            #pragma unroll
            for (int msk = 8; msk >= 1; msk >>= 1)
                sum += __shfl_xor(sum, msk, 32);
            float inv = 1.f / sum;
            #pragma unroll
            for (int nt = 0; nt < 4; ++nt) acc[mt][nt][e] *= inv;
        }
    // D-layout -> row-major bf16 in LDS so it can be reloaded as an A fragment
    #pragma unroll
    for (int mt = 0; mt < 4; ++mt)
        #pragma unroll
        for (int nt = 0; nt < 4; ++nt)
            #pragma unroll
            for (int e = 0; e < 8; ++e)
                att[(mt*16 + e + 8*lh)*64 + nt*16 + cn] = f2bf(acc[mt][nt][e]);
    // attn @ V : K = 64 (2 steps), N = 32 (2 tiles), B = vT (NxK transposed)
    const bf16* vb = vT + (size_t)(win*8 + head)*2048;    // [32][64]
    floatx8 oacc[4][2] = {};
    #pragma unroll
    for (int ks = 0; ks < 2; ++ks) {
        bf16x16 b2[2];
        #pragma unroll
        for (int nt = 0; nt < 2; ++nt) b2[nt] = load_frag(vb, nt*16, 64, ks*32, lane);
        #pragma unroll
        for (int mt = 0; mt < 4; ++mt) {
            bf16x16 a2 = load_frag(att, mt*16, 64, ks*32, lane);
            #pragma unroll
            for (int nt = 0; nt < 2; ++nt)
                oacc[mt][nt] = wmma_bf16(a2, b2[nt], oacc[mt][nt]);
        }
    }
    #pragma unroll
    for (int mt = 0; mt < 4; ++mt)
        #pragma unroll
        for (int nt = 0; nt < 2; ++nt)
            #pragma unroll
            for (int e = 0; e < 8; ++e) {
                int n = mt*16 + e + 8*lh, d = nt*16 + cn;
                o[(size_t)win*16384 + n*256 + head*32 + d] = f2bf(oacc[mt][nt][e]);
            }
}

// ---------- K4: out-proj + window reverse + un-shift + residual ----------
__global__ void k_oproj(const bf16* __restrict__ o, const bf16* __restrict__ woT,
                        const float* __restrict__ bo, const float* __restrict__ x,
                        float* __restrict__ x2) {
    int wave = threadIdx.x >> 5, lane = threadIdx.x & 31;
    int win = blockIdx.x;
    int cn = lane & 15, lh = lane >> 4;
    int b = win >> 8, wy = (win >> 4) & 15, wx = win & 15;
    const bf16* A = o + (size_t)win * 16384;
    for (int s = 0; s < 2; ++s) {
        int nbase = (wave*2 + s) * 16;
        floatx8 acc[4] = {};
        for (int kb = 0; kb < 256; kb += 32) {
            bf16x16 bfr = load_frag(woT + (size_t)nbase*256, 0, 256, kb, lane);
            #pragma unroll
            for (int mt = 0; mt < 4; ++mt)
                acc[mt] = wmma_bf16(load_frag(A, mt*16, 256, kb, lane), bfr, acc[mt]);
        }
        int c = nbase + cn; float bias = bo[c];
        #pragma unroll
        for (int mt = 0; mt < 4; ++mt)
            #pragma unroll
            for (int e = 0; e < 8; ++e) {
                int n = mt*16 + e + 8*lh;
                int hs = wy*8 + (n >> 3), wsx = wx*8 + (n & 7);
                int h = (hs + 4) & 127, w = (wsx + 4) & 127;
                size_t idx = ((size_t)b << 14) + h*128 + w;
                x2[idx*256 + c] = x[idx*256 + c] + acc[mt][e] + bias;
            }
    }
}

// ---------- K5: LN2 + cast ----------
__global__ void k_ln2(const float* __restrict__ x2, const float* __restrict__ g,
                      const float* __restrict__ be, bf16* __restrict__ y) {
    int wave = threadIdx.x >> 5, lane = threadIdx.x & 31;
    int t = blockIdx.x * 8 + wave;
    const float* row = x2 + (size_t)t * 256 + lane * 8;
    float v[8];
    float4 p0 = *reinterpret_cast<const float4*>(row);
    float4 p1 = *reinterpret_cast<const float4*>(row + 4);
    v[0]=p0.x; v[1]=p0.y; v[2]=p0.z; v[3]=p0.w;
    v[4]=p1.x; v[5]=p1.y; v[6]=p1.z; v[7]=p1.w;
    float s = 0.f, s2 = 0.f;
    #pragma unroll
    for (int j = 0; j < 8; ++j) { s += v[j]; s2 += v[j]*v[j]; }
    #pragma unroll
    for (int m = 16; m >= 1; m >>= 1) {
        s  += __shfl_xor(s,  m, 32);
        s2 += __shfl_xor(s2, m, 32);
    }
    float mu = s * (1.f/256.f);
    float rstd = rsqrtf(s2*(1.f/256.f) - mu*mu + 1e-5f);
    union { bf16x8 vec; bf16 e[8]; } o;
    #pragma unroll
    for (int j = 0; j < 8; ++j)
        o.e[j] = f2bf((v[j] - mu) * rstd * g[lane*8+j] + be[lane*8+j]);
    *reinterpret_cast<bf16x8*>(y + (size_t)t * 256 + lane * 8) = o.vec;
}

// ---------- K6: MLP1 (256->1024) + GELU ----------
__global__ void k_mlp1(const bf16* __restrict__ y, const bf16* __restrict__ w1T,
                       const float* __restrict__ bl1, bf16* __restrict__ h1) {
    int wave = threadIdx.x >> 5, lane = threadIdx.x & 31;
    int mb = blockIdx.x >> 3, nb = blockIdx.x & 7;
    int nbase = nb*128 + wave*16;
    int cn = lane & 15, lh = lane >> 4;
    const bf16* A = y + (size_t)mb * 64 * 256;
    floatx8 acc[4] = {};
    for (int kb = 0; kb < 256; kb += 32) {
        bf16x16 bfr = load_frag(w1T + (size_t)nbase*256, 0, 256, kb, lane);
        #pragma unroll
        for (int mt = 0; mt < 4; ++mt)
            acc[mt] = wmma_bf16(load_frag(A, mt*16, 256, kb, lane), bfr, acc[mt]);
    }
    int c = nbase + cn; float bias = bl1[c];
    #pragma unroll
    for (int mt = 0; mt < 4; ++mt)
        #pragma unroll
        for (int e = 0; e < 8; ++e) {
            int row = mb*64 + mt*16 + e + 8*lh;
            h1[(size_t)row*1024 + c] = f2bf(geluf(acc[mt][e] + bias));
        }
}

// ---------- K7: depthwise 3x3 conv + bias + GELU ----------
__global__ void k_dwconv(const bf16* __restrict__ h1, const float* __restrict__ dww,
                         const float* __restrict__ dwb, bf16* __restrict__ h2) {
    int tid = blockIdx.x * 256 + threadIdx.x;       // 8,388,608 threads
    int pix = tid >> 7, cb = tid & 127;
    int c0 = cb * 8;
    int b = pix >> 14, hw = pix & 16383, hh = hw >> 7, ww = hw & 127;
    float acc[8] = {};
    #pragma unroll
    for (int dy = -1; dy <= 1; ++dy) {
        int y = hh + dy;
        if (y < 0 || y > 127) continue;
        #pragma unroll
        for (int dx = -1; dx <= 1; ++dx) {
            int xp = ww + dx;
            if (xp < 0 || xp > 127) continue;
            const bf16* p = h1 + ((size_t)(b << 14) + y*128 + xp)*1024 + c0;
            bf16x8 vv = *reinterpret_cast<const bf16x8*>(p);
            const float* wp = dww + ((dy+1)*3 + (dx+1))*1024 + c0;
            #pragma unroll
            for (int j = 0; j < 8; ++j) acc[j] += bf2f(vv[j]) * wp[j];
        }
    }
    union { bf16x8 vec; bf16 e[8]; } o;
    #pragma unroll
    for (int j = 0; j < 8; ++j) o.e[j] = f2bf(geluf(acc[j] + dwb[c0 + j]));
    *reinterpret_cast<bf16x8*>(h2 + (size_t)pix*1024 + c0) = o.vec;
}

// ---------- K8: MLP2 (1024->256) + residual -> f32 output ----------
// A-tiles (64x32 bf16 = 4KB) are staged into LDS with async global->LDS copies
// (double buffered: transfer of step kb+1 overlaps the 8 WMMAs of step kb) and
// shared by all 8 waves. Each wave keeps both of its N-strips' accumulators
// live so each LDS A-fragment feeds 2 WMMAs.
__global__ void k_mlp2(const bf16* __restrict__ h2, const bf16* __restrict__ w2T,
                       const float* __restrict__ bl2, const float* __restrict__ x2,
                       float* __restrict__ out) {
    __shared__ bf16 atile[2][64 * 32];              // 2 x 4KB
    int tid = threadIdx.x;
    int wave = tid >> 5, lane = tid & 31;
    int mb = blockIdx.x;
    int cn = lane & 15, lh = lane >> 4;
    const bf16* A = h2 + (size_t)mb * 64 * 1024;

    // staging map: thread t moves 16B; row = t>>2, 16B-quarter = t&3
    const bf16* gsrc = A + (size_t)(tid >> 2) * 1024 + (tid & 3) * 8;
    unsigned lds_base0 = (unsigned)(size_t)(&atile[0][0]) + tid * 16;
    unsigned lds_base1 = (unsigned)(size_t)(&atile[1][0]) + tid * 16;

    async_load_b128(lds_base0, gsrc);               // stage kb = 0
    wait_asynccnt0();
    __syncthreads();

    int nbase = wave * 32;                          // strips nbase, nbase+16
    floatx8 acc[2][4] = {};
    for (int kb = 0; kb < 32; ++kb) {
        if (kb + 1 < 32)                            // stage kb+1 into other buffer
            async_load_b128((kb & 1) ? lds_base0 : lds_base1, gsrc + (kb + 1) * 32);
        const bf16* at = &atile[kb & 1][0];
        bf16x16 bfr0 = load_frag(w2T + (size_t)nbase*1024,      0, 1024, kb*32, lane);
        bf16x16 bfr1 = load_frag(w2T + (size_t)(nbase+16)*1024, 0, 1024, kb*32, lane);
        #pragma unroll
        for (int mt = 0; mt < 4; ++mt) {
            bf16x16 afr = load_frag(at, mt*16, 32, 0, lane);
            acc[0][mt] = wmma_bf16(afr, bfr0, acc[0][mt]);
            acc[1][mt] = wmma_bf16(afr, bfr1, acc[1][mt]);
        }
        if (kb + 1 < 32) {
            wait_asynccnt0();                       // my slice of kb+1 landed
            __syncthreads();                        // all slices landed, all readers done
        }
    }
    #pragma unroll
    for (int s = 0; s < 2; ++s) {
        int c = nbase + s*16 + cn; float bias = bl2[c];
        #pragma unroll
        for (int mt = 0; mt < 4; ++mt)
            #pragma unroll
            for (int e = 0; e < 8; ++e) {
                size_t row = (size_t)mb*64 + mt*16 + e + 8*lh;
                out[row*256 + c] = acc[s][mt][e] + bias + x2[row*256 + c];
            }
    }
}

extern "C" void kernel_launch(void* const* d_in, const int* in_sizes, int n_in,
                              void* d_out, int out_size, void* d_ws, size_t ws_size,
                              hipStream_t stream) {
    (void)in_sizes; (void)n_in; (void)out_size; (void)ws_size;
    const float* x   = (const float*)d_in[0];
    const float* g1  = (const float*)d_in[1];
    const float* be1 = (const float*)d_in[2];
    const float* wq  = (const float*)d_in[3];
    const float* bq  = (const float*)d_in[4];
    const float* wkv = (const float*)d_in[5];
    const float* bkv = (const float*)d_in[6];
    const float* rpb = (const float*)d_in[7];
    const float* wo  = (const float*)d_in[8];
    const float* bo  = (const float*)d_in[9];
    const float* g2  = (const float*)d_in[10];
    const float* be2 = (const float*)d_in[11];
    const float* w1  = (const float*)d_in[12];
    const float* bl1 = (const float*)d_in[13];
    const float* dww = (const float*)d_in[14];
    const float* dwb = (const float*)d_in[15];
    const float* w2  = (const float*)d_in[16];
    const float* bl2 = (const float*)d_in[17];
    float* out = (float*)d_out;

    char* ws = (char*)d_ws;
    const size_t MB = 1ull << 20;
    bf16* wqT  = (bf16*)(ws + 0);          // 256x256
    bf16* wkvT = (bf16*)(ws + 131072);     // 512x256
    bf16* woT  = (bf16*)(ws + 393216);     // 256x256
    bf16* w1T  = (bf16*)(ws + 524288);     // 1024x256
    bf16* w2T  = (bf16*)(ws + 1048576);    // 256x1024
    float* x2  = (float*)(ws + 2*MB);      // 64 MB
    bf16* ybf  = (bf16*)(ws + 66*MB);      // 32 MB
    bf16* xw   = (bf16*)(ws + 98*MB);      // 32 MB
    bf16* qb   = (bf16*)(ws + 130*MB);     // 32 MB
    bf16* kb   = (bf16*)(ws + 162*MB);     // 32 MB
    bf16* vT   = (bf16*)(ws + 194*MB);     // 32 MB
    bf16* ob   = (bf16*)(ws + 226*MB);     // 32 MB (ends 258 MB)
    bf16* h1   = (bf16*)(ws + 130*MB);     // 128 MB, aliases q/k/vT/ob (dead by then)
    bf16* h2b  = (bf16*)(ws + 258*MB);     // 128 MB (ends 386 MB)

    k_wt<<<256,  256, 0, stream>>>(wq,  wqT,  256, 256);
    k_wt<<<512,  256, 0, stream>>>(wkv, wkvT, 256, 512);
    k_wt<<<256,  256, 0, stream>>>(wo,  woT,  256, 256);
    k_wt<<<1024, 256, 0, stream>>>(w1,  w1T,  256, 1024);
    k_wt<<<1024, 256, 0, stream>>>(w2,  w2T,  1024, 256);

    k_ln_shift_part<<<8192, 256, 0, stream>>>(x, g1, be1, xw);
    k_qkv<<<6144, 256, 0, stream>>>(xw, wqT, wkvT, bq, bkv, qb, kb, vT);
    k_attn<<<8192, 32, 0, stream>>>(qb, kb, vT, rpb, ob);
    k_oproj<<<1024, 256, 0, stream>>>(ob, woT, bo, x, x2);
    k_ln2<<<8192, 256, 0, stream>>>(x2, g2, be2, ybf);
    k_mlp1<<<8192, 256, 0, stream>>>(ybf, w1T, bl1, h1);
    k_dwconv<<<32768, 256, 0, stream>>>(h1, dww, dwb, h2b);
    k_mlp2<<<1024, 256, 0, stream>>>(h2b, w2T, bl2, x2, out);
}
